// SegMap_pooling_v0_89558658056315
// MI455X (gfx1250) — matrix operand order repaired and checked by reading, not compiled, not run on hardware
//
#include <hip/hip_runtime.h>

// ---------------- problem constants ----------------
#define N_PTS 120000
#define D_DIM 256
#define BATCH 2
#define HEADS 8
#define DH 32
#define OUTC 20
// query grid: ceil(480/8)=60, ceil(360/8)=45, ceil(32/8)=4
#define ZD 60
#define YD 45
#define XD 4
#define GQ (ZD * YD * XD)  // 10800
#define SQ (BATCH * GQ)    // 21600
// embed grid: ceil(480/40)=12, ceil(360/15)=24, ceil(32/16)=2
#define ZE 12
#define YE 24
#define XE 2
#define GK (ZE * YE * XE)  // 576
#define SK (BATCH * GK)    // 1152
#define QTILES (GQ / 16)   // 675
#define KTILES (GK / 16)   // 36

typedef unsigned short u16t;
typedef __attribute__((ext_vector_type(16))) __bf16 v16bf;
typedef __attribute__((ext_vector_type(8))) float v8f;

union ABf {
  v16bf v;
  u16t u[16];
  uint4 q[2];
};

// ---------------- helpers ----------------
__device__ inline u16t f2bf(float f) {
  unsigned int u = __float_as_uint(f);
  unsigned int r = u + 0x7FFFu + ((u >> 16) & 1u);  // round-nearest-even
  return (u16t)(r >> 16);
}
__device__ inline float bf2f(u16t h) { return __uint_as_float(((unsigned int)h) << 16); }

__device__ inline void atomicMaxF32(float* addr, float val) {
  if (val >= 0.f)
    atomicMax((int*)addr, __float_as_int(val));
  else
    atomicMin((unsigned int*)addr, __float_as_uint(val));
}

// one channel of the sine positional encoding (nf = 42, 6*42 = 252, pad to 256)
__device__ inline float pe_ch(float cz, float cy, float cx, int ch) {
  if (ch >= 252) return 0.f;
  int axis = ch / 84;
  int w = ch % 84;
  int f = (w >= 42) ? (w - 42) : w;
  float c = (axis == 0) ? cz : ((axis == 1) ? cy : cx);
  float sp = (axis == 0) ? 480.f : ((axis == 1) ? 360.f : 32.f);
  float cn = c / sp * 6.2831853071795864f;
  float t = __powf(10000.f, (float)f * (1.f / 42.f));
  float ang = cn / t;
  return (w >= 42) ? __cosf(ang) : __sinf(ang);
}

// ---------------- tiny utility kernels ----------------
__global__ __launch_bounds__(256) void fill_f32(float* p, float v, long count) {
  long i = (long)blockIdx.x * blockDim.x + threadIdx.x;
  if (i < count) p[i] = v;
}

__global__ __launch_bounds__(256) void f32_to_bf16(const float* __restrict__ src,
                                                   u16t* __restrict__ dst, long count) {
  long i = (long)blockIdx.x * blockDim.x + threadIdx.x;
  if (i < count) dst[i] = f2bf(src[i]);
}

// WT[c*256 + k] = bf16(W[k*256 + c])
__global__ __launch_bounds__(256) void transpose_w(const float* __restrict__ W,
                                                   u16t* __restrict__ WT) {
  int id = blockIdx.x * blockDim.x + threadIdx.x;  // 65536 threads
  int k = id >> 8, c = id & 255;
  WT[c * D_DIM + k] = f2bf(W[k * D_DIM + c]);
}

// per-point segment ids + occupancy counts
__global__ __launch_bounds__(256) void prep_seg(const int* __restrict__ indices,
                                                int* __restrict__ seg_q, int* __restrict__ seg_k,
                                                int* __restrict__ cnt_q, int* __restrict__ cnt_k) {
  long n = (long)blockIdx.x * blockDim.x + threadIdx.x;
  if (n >= N_PTS) return;
  const int* ip = indices + n * 4;
  int b = ip[0], z = ip[1], y = ip[2], x = ip[3];
  int sq = ((b * ZD + z / 8) * YD + y / 8) * XD + x / 8;
  int sk = ((b * ZE + z / 40) * YE + y / 15) * XE + x / 16;
  seg_q[n] = sq;
  seg_k[n] = sk;
  atomicAdd(&cnt_q[sq], 1);
  atomicAdd(&cnt_k[sk], 1);
}

// ---------------- WMMA GEMM core: rows [rb,rb+16) x cols [colbase,colbase+128) ----------------
__device__ inline void gemm_core(const u16t* __restrict__ A, const u16t* __restrict__ WT, long M,
                                 long rb, int colbase, int lrow, int hi, v8f* acc) {
  long ar = rb + lrow;
  if (ar >= M) ar = M - 1;  // clamp; out-of-range rows discarded at epilogue
  const u16t* arow = A + ar * D_DIM;
  for (int k0 = 0; k0 < D_DIM; k0 += 32) {
    ABf a;
    const u16t* ap = arow + k0 + hi * 8;
    a.q[0] = *(const uint4*)ap;
    a.q[1] = *(const uint4*)(ap + 16);
#pragma unroll
    for (int t = 0; t < 8; ++t) {
      int c = colbase + t * 16 + lrow;
      const u16t* bp = WT + (long)c * D_DIM + k0 + hi * 8;
      ABf b;
      b.q[0] = *(const uint4*)bp;
      b.q[1] = *(const uint4*)(bp + 16);
      acc[t] = __builtin_amdgcn_wmma_f32_16x16x32_bf16(false, a.v, false, b.v, (short)0, acc[t],
                                                       false, false);
    }
  }
}

// GEMM + bias, epilogue = atomic segment-max scatter
__global__ __launch_bounds__(256) void gemm_scatter_max(const u16t* __restrict__ A,
                                                        const u16t* __restrict__ WT,
                                                        const float* __restrict__ bias,
                                                        const int* __restrict__ seg,
                                                        float* __restrict__ outmax, long M) {
  int wave = threadIdx.x >> 5, lane = threadIdx.x & 31;
  int wr = wave & 3, wc = wave >> 2;
  int lrow = lane & 15, hi = lane >> 4;
  long rb = (long)blockIdx.x * 64 + wr * 16;
  int colbase = wc * 128;
  v8f acc[8] = {};
  gemm_core(A, WT, M, rb, colbase, lrow, hi, acc);
#pragma unroll
  for (int t = 0; t < 8; ++t) {
    int col = colbase + t * 16 + lrow;
    float bcol = bias[col];
#pragma unroll
    for (int r = 0; r < 8; ++r) {
      long n = rb + r + 8 * hi;
      if (n < M) {
        int s = seg[n];
        atomicMaxF32(&outmax[(long)s * D_DIM + col], acc[t][r] + bcol);
      }
    }
  }
}

// recompute GEMM deterministically; where value == segment max, record max point index
__global__ __launch_bounds__(256) void gemm_winner(const u16t* __restrict__ A,
                                                   const u16t* __restrict__ WT,
                                                   const float* __restrict__ bias,
                                                   const int* __restrict__ seg,
                                                   const float* __restrict__ emax,
                                                   int* __restrict__ idxw, long M) {
  int wave = threadIdx.x >> 5, lane = threadIdx.x & 31;
  int wr = wave & 3, wc = wave >> 2;
  int lrow = lane & 15, hi = lane >> 4;
  long rb = (long)blockIdx.x * 64 + wr * 16;
  int colbase = wc * 128;
  v8f acc[8] = {};
  gemm_core(A, WT, M, rb, colbase, lrow, hi, acc);
#pragma unroll
  for (int t = 0; t < 8; ++t) {
    int col = colbase + t * 16 + lrow;
    float bcol = bias[col];
#pragma unroll
    for (int r = 0; r < 8; ++r) {
      long n = rb + r + 8 * hi;
      if (n < M) {
        int s = seg[n];
        float v = acc[t][r] + bcol;
        if (v == emax[(long)s * D_DIM + col]) atomicMax(&idxw[(long)s * D_DIM + col], (int)n);
      }
    }
  }
}

// GEMM + bias -> optional f32 out and/or bf16 out
__global__ __launch_bounds__(256) void gemm_bias(const u16t* __restrict__ A,
                                                 const u16t* __restrict__ WT,
                                                 const float* __restrict__ bias,
                                                 float* __restrict__ outf, u16t* __restrict__ outb,
                                                 long M) {
  int wave = threadIdx.x >> 5, lane = threadIdx.x & 31;
  int wr = wave & 3, wc = wave >> 2;
  int lrow = lane & 15, hi = lane >> 4;
  long rb = (long)blockIdx.x * 64 + wr * 16;
  int colbase = wc * 128;
  v8f acc[8] = {};
  gemm_core(A, WT, M, rb, colbase, lrow, hi, acc);
#pragma unroll
  for (int t = 0; t < 8; ++t) {
    int col = colbase + t * 16 + lrow;
    float bcol = bias[col];
#pragma unroll
    for (int r = 0; r < 8; ++r) {
      long n = rb + r + 8 * hi;
      if (n < M) {
        float v = acc[t][r] + bcol;
        if (outf) outf[n * D_DIM + col] = v;
        if (outb) outb[n * D_DIM + col] = f2bf(v);
      }
    }
  }
}

// ---------------- embedding winner coords: mean over channels of de[idx] * EP ----------------
__global__ __launch_bounds__(256) void emb_coords_kernel(const int* __restrict__ idxw,
                                                         const int* __restrict__ indices,
                                                         float* __restrict__ emb_c) {
  __shared__ float rz[256], ry[256], rx[256];
  int s = blockIdx.x, t = threadIdx.x;
  int id = idxw[(long)s * D_DIM + t];
  id = (id < 0) ? 0 : ((id > N_PTS - 1) ? N_PTS - 1 : id);
  const int* ip = indices + (long)id * 4;
  rz[t] = (float)(ip[1] / 40);
  ry[t] = (float)(ip[2] / 15);
  rx[t] = (float)(ip[3] / 16);
  __syncthreads();
  for (int st = 128; st > 0; st >>= 1) {
    if (t < st) {
      rz[t] += rz[t + st];
      ry[t] += ry[t + st];
      rx[t] += rx[t + st];
    }
    __syncthreads();
  }
  if (t == 0) {
    emb_c[s * 3 + 0] = rz[0] * (1.f / 256.f) * 40.f;
    emb_c[s * 3 + 1] = ry[0] * (1.f / 256.f) * 15.f;
    emb_c[s * 3 + 2] = rx[0] * (1.f / 256.f) * 16.f;
  }
}

// pool branch: mask empties to 0, add sine PE, emit f32 + bf16
__global__ __launch_bounds__(256) void finalize_pool(const float* __restrict__ pool_max,
                                                     const int* __restrict__ cnt_q,
                                                     float* __restrict__ pool_pe,
                                                     u16t* __restrict__ pool_bf) {
  long id = (long)blockIdx.x * blockDim.x + threadIdx.x;
  if (id >= (long)SQ * D_DIM) return;
  int s = (int)(id >> 8), ch = (int)(id & 255);
  int rem = s % GQ;
  float cz = (float)((rem / (YD * XD)) * 8);
  float cy = (float)(((rem / XD) % YD) * 8);
  float cx = (float)((rem % XD) * 8);
  float v = (cnt_q[s] > 0) ? pool_max[id] : 0.f;
  v += pe_ch(cz, cy, cx, ch);
  pool_pe[id] = v;
  pool_bf[id] = f2bf(v);
}

// emb branch: mask empties, add sine PE at winner coords, emit bf16 (feeds K/V proj)
__global__ __launch_bounds__(256) void finalize_emb(const float* __restrict__ emb_max,
                                                    const int* __restrict__ cnt_k,
                                                    const float* __restrict__ emb_c,
                                                    u16t* __restrict__ emb_bf) {
  long id = (long)blockIdx.x * blockDim.x + threadIdx.x;
  if (id >= (long)SK * D_DIM) return;
  int s = (int)(id >> 8), ch = (int)(id & 255);
  float v = (cnt_k[s] > 0) ? emb_max[id] : 0.f;
  v += pe_ch(emb_c[s * 3 + 0], emb_c[s * 3 + 1], emb_c[s * 3 + 2], ch);
  emb_bf[id] = f2bf(v);
}

// vT[(b*H+h)*32*576 + d*576 + key] = v_bf[(b*GK+key)*256 + h*32 + d]
__global__ __launch_bounds__(256) void transpose_v(const u16t* __restrict__ vbf,
                                                   u16t* __restrict__ vT) {
  long id = (long)blockIdx.x * blockDim.x + threadIdx.x;
  if (id >= (long)SK * D_DIM) return;
  int b = (int)(id / ((long)GK * D_DIM));
  int rem = (int)(id % ((long)GK * D_DIM));
  int key = rem >> 8, c = rem & 255;
  int h = c >> 5, d = c & 31;
  vT[((long)(b * HEADS + h) * DH + d) * GK + key] = vbf[id];
}

// ---------------- flash-style attention: 1 wave per (b, h, 16-query tile) ----------------
__global__ __launch_bounds__(32) void attn_kernel(const u16t* __restrict__ qbf,
                                                  const u16t* __restrict__ kbf,
                                                  const u16t* __restrict__ vT,
                                                  const int* __restrict__ cnt_k,
                                                  u16t* __restrict__ attn_out) {
  __shared__ __align__(16) float sc[16 * GK];
  __shared__ __align__(16) u16t pb[16 * GK];
  int blk = blockIdx.x;
  int qt = blk % QTILES;
  int h = (blk / QTILES) % HEADS;
  int b = blk / (QTILES * HEADS);
  int lane = threadIdx.x;
  int lrow = lane & 15, hi = lane >> 4;
  long qrow0 = (long)b * GQ + (long)qt * 16;

  // Q tile as WMMA A operand
  ABf a;
  {
    const u16t* ap = qbf + (qrow0 + lrow) * D_DIM + h * DH + hi * 8;
    a.q[0] = *(const uint4*)ap;
    a.q[1] = *(const uint4*)(ap + 16);
  }
  const float scale = 0.1767766952966369f;  // 1/sqrt(32)

  for (int kt = 0; kt < KTILES; ++kt) {
    int key = kt * 16 + lrow;
    const u16t* bp = kbf + ((long)b * GK + key) * D_DIM + h * DH + hi * 8;
    ABf bb;
    bb.q[0] = *(const uint4*)bp;
    bb.q[1] = *(const uint4*)(bp + 16);
    v8f s = {};
    s = __builtin_amdgcn_wmma_f32_16x16x32_bf16(false, a.v, false, bb.v, (short)0, s, false, false);
    float m = (cnt_k[b * GK + key] > 0) ? 0.f : -1e9f;  // mask by key column
#pragma unroll
    for (int r = 0; r < 8; ++r) sc[(r + 8 * hi) * GK + kt * 16 + lrow] = s[r] * scale + m;
  }
  __syncthreads();

  // softmax: lane pair (lrow, hi) handles half of row lrow
  {
    int row = lrow;
    float mx = -1e30f;
    for (int j = hi * 288; j < hi * 288 + 288; ++j) mx = fmaxf(mx, sc[row * GK + j]);
    mx = fmaxf(mx, __shfl_xor(mx, 16, 32));
    float sum = 0.f;
    for (int j = hi * 288; j < hi * 288 + 288; ++j) {
      float e = __expf(sc[row * GK + j] - mx);
      sum += e;
      sc[row * GK + j] = e;
    }
    sum += __shfl_xor(sum, 16, 32);
    float inv = 1.f / sum;
    for (int j = hi * 288; j < hi * 288 + 288; ++j) pb[row * GK + j] = f2bf(sc[row * GK + j] * inv);
  }
  __syncthreads();

  // out[16,32] = P[16,576] @ V[576,32] via bf16 WMMA, K-chunks of 32
  const u16t* vbase = vT + (long)(b * HEADS + h) * DH * GK;
  v8f o0 = {}, o1 = {};
  for (int kc = 0; kc < GK; kc += 32) {
    ABf pa;
    const u16t* pp = &pb[lrow * GK + kc + hi * 8];
    pa.q[0] = *(const uint4*)pp;
    pa.q[1] = *(const uint4*)(pp + 16);
    const u16t* b0 = vbase + (long)lrow * GK + kc + hi * 8;
    const u16t* b1 = vbase + (long)(16 + lrow) * GK + kc + hi * 8;
    ABf vb0, vb1;
    vb0.q[0] = *(const uint4*)b0;
    vb0.q[1] = *(const uint4*)(b0 + 16);
    vb1.q[0] = *(const uint4*)b1;
    vb1.q[1] = *(const uint4*)(b1 + 16);
    o0 = __builtin_amdgcn_wmma_f32_16x16x32_bf16(false, pa.v, false, vb0.v, (short)0, o0, false,
                                                 false);
    o1 = __builtin_amdgcn_wmma_f32_16x16x32_bf16(false, pa.v, false, vb1.v, (short)0, o1, false,
                                                 false);
  }
#pragma unroll
  for (int r = 0; r < 8; ++r) {
    long rr = qrow0 + r + 8 * hi;
    attn_out[rr * D_DIM + h * DH + lrow] = f2bf(o0[r]);
    attn_out[rr * D_DIM + h * DH + 16 + lrow] = f2bf(o1[r]);
  }
}

// ---------------- residual + LayerNorm (one block per row) ----------------
__global__ __launch_bounds__(256) void residual_ln(const float* __restrict__ qin,
                                                   const float* __restrict__ o,
                                                   const float* __restrict__ g,
                                                   const float* __restrict__ bb,
                                                   float* __restrict__ outp) {
  __shared__ float red[256];
  long row = blockIdx.x;
  int t = threadIdx.x;
  float v = qin[row * D_DIM + t] + o[row * D_DIM + t];
  red[t] = v;
  __syncthreads();
  for (int st = 128; st > 0; st >>= 1) {
    if (t < st) red[t] += red[t + st];
    __syncthreads();
  }
  float mean = red[0] * (1.f / 256.f);
  __syncthreads();
  float d = v - mean;
  red[t] = d * d;
  __syncthreads();
  for (int st = 128; st > 0; st >>= 1) {
    if (t < st) red[t] += red[t + st];
    __syncthreads();
  }
  float var = red[0] * (1.f / 256.f);
  outp[row * D_DIM + t] = d * rsqrtf(var + 1e-5f) * g[t] + bb[t];
}

// ---------------- seg head: logits = (features + mhca[seg_q]) @ W_seg + b_seg ----------------
__global__ __launch_bounds__(256) void seg_head(const float* __restrict__ feat,
                                                const float* __restrict__ mhca,
                                                const int* __restrict__ seg,
                                                const float* __restrict__ Wseg,
                                                const float* __restrict__ bseg,
                                                float* __restrict__ logits) {
  __shared__ float sf[256];
  long n = blockIdx.x;
  int t = threadIdx.x;
  sf[t] = feat[n * D_DIM + t] + mhca[(long)seg[n] * D_DIM + t];
  __syncthreads();
  if (t < OUTC) {
    float acc = bseg[t];
    for (int k = 0; k < D_DIM; ++k) acc += sf[k] * Wseg[k * OUTC + t];
    logits[n * OUTC + t] = acc;
  }
}

// ---------------- host launcher ----------------
extern "C" void kernel_launch(void* const* d_in, const int* in_sizes, int n_in, void* d_out,
                              int out_size, void* d_ws, size_t ws_size, hipStream_t stream) {
  (void)in_sizes;
  (void)n_in;
  (void)out_size;
  (void)ws_size;
  const float* features = (const float*)d_in[0];
  const float* W_pool = (const float*)d_in[1];
  const float* b_pool = (const float*)d_in[2];
  const float* W_emb = (const float*)d_in[3];
  const float* b_emb = (const float*)d_in[4];
  const float* Wq = (const float*)d_in[5];
  const float* bq = (const float*)d_in[6];
  const float* Wk = (const float*)d_in[7];
  const float* bk = (const float*)d_in[8];
  const float* Wv = (const float*)d_in[9];
  const float* bv = (const float*)d_in[10];
  const float* Wo = (const float*)d_in[11];
  const float* bo = (const float*)d_in[12];
  const float* ln_g = (const float*)d_in[13];
  const float* ln_b = (const float*)d_in[14];
  const float* W_seg = (const float*)d_in[15];
  const float* b_seg = (const float*)d_in[16];
  const int* indices = (const int*)d_in[17];
  float* logits = (float*)d_out;

  char* ws = (char*)d_ws;
  size_t off = 0;
  auto alloc = [&](size_t bytes) -> void* {
    void* p = ws + off;
    off += (bytes + 255) & ~(size_t)255;
    return p;
  };
  u16t* featbf = (u16t*)alloc((size_t)N_PTS * D_DIM * 2);
  u16t* WTp = (u16t*)alloc((size_t)D_DIM * D_DIM * 2);
  u16t* WTe = (u16t*)alloc((size_t)D_DIM * D_DIM * 2);
  u16t* WTq = (u16t*)alloc((size_t)D_DIM * D_DIM * 2);
  u16t* WTk = (u16t*)alloc((size_t)D_DIM * D_DIM * 2);
  u16t* WTv = (u16t*)alloc((size_t)D_DIM * D_DIM * 2);
  u16t* WTo = (u16t*)alloc((size_t)D_DIM * D_DIM * 2);
  int* seg_q = (int*)alloc((size_t)N_PTS * 4);
  int* seg_k = (int*)alloc((size_t)N_PTS * 4);
  int* cnt_q = (int*)alloc((size_t)SQ * 4);
  int* cnt_k = (int*)alloc((size_t)SK * 4);
  float* pool_max = (float*)alloc((size_t)SQ * D_DIM * 4);
  float* emb_max = (float*)alloc((size_t)SK * D_DIM * 4);
  int* idxw = (int*)alloc((size_t)SK * D_DIM * 4);
  float* emb_c = (float*)alloc((size_t)SK * 3 * 4);
  float* pool_pe = (float*)alloc((size_t)SQ * D_DIM * 4);
  u16t* pool_bf = (u16t*)alloc((size_t)SQ * D_DIM * 2);
  u16t* emb_bf = (u16t*)alloc((size_t)SK * D_DIM * 2);
  u16t* qbf = (u16t*)alloc((size_t)SQ * D_DIM * 2);
  u16t* kbf = (u16t*)alloc((size_t)SK * D_DIM * 2);
  u16t* vbf = (u16t*)alloc((size_t)SK * D_DIM * 2);
  u16t* vT = (u16t*)alloc((size_t)BATCH * HEADS * DH * GK * 2);
  u16t* attnbf = (u16t*)alloc((size_t)SQ * D_DIM * 2);
  float* o32 = (float*)alloc((size_t)SQ * D_DIM * 4);
  float* mhca = (float*)alloc((size_t)SQ * D_DIM * 4);

  // init
  hipMemsetAsync(cnt_q, 0, (size_t)SQ * 4, stream);
  hipMemsetAsync(cnt_k, 0, (size_t)SK * 4, stream);
  hipMemsetAsync(idxw, 0xFF, (size_t)SK * D_DIM * 4, stream);  // -1
  {
    long c = (long)SQ * D_DIM;
    fill_f32<<<(int)((c + 255) / 256), 256, 0, stream>>>(pool_max, -INFINITY, c);
  }
  {
    long c = (long)SK * D_DIM;
    fill_f32<<<(int)((c + 255) / 256), 256, 0, stream>>>(emb_max, -INFINITY, c);
  }

  // prep
  prep_seg<<<(N_PTS + 255) / 256, 256, 0, stream>>>(indices, seg_q, seg_k, cnt_q, cnt_k);
  {
    long c = (long)N_PTS * D_DIM;
    f32_to_bf16<<<(int)((c + 255) / 256), 256, 0, stream>>>(features, featbf, c);
  }
  transpose_w<<<256, 256, 0, stream>>>(W_pool, WTp);
  transpose_w<<<256, 256, 0, stream>>>(W_emb, WTe);
  transpose_w<<<256, 256, 0, stream>>>(Wq, WTq);
  transpose_w<<<256, 256, 0, stream>>>(Wk, WTk);
  transpose_w<<<256, 256, 0, stream>>>(Wv, WTv);
  transpose_w<<<256, 256, 0, stream>>>(Wo, WTo);

  // big GEMMs with fused scatter-max, then winner-index pass
  int nblkN = (N_PTS + 63) / 64;
  gemm_scatter_max<<<nblkN, 256, 0, stream>>>(featbf, WTp, b_pool, seg_q, pool_max, N_PTS);
  gemm_scatter_max<<<nblkN, 256, 0, stream>>>(featbf, WTe, b_emb, seg_k, emb_max, N_PTS);
  gemm_winner<<<nblkN, 256, 0, stream>>>(featbf, WTe, b_emb, seg_k, emb_max, idxw, N_PTS);

  emb_coords_kernel<<<SK, 256, 0, stream>>>(idxw, indices, emb_c);
  {
    long c = (long)SQ * D_DIM;
    finalize_pool<<<(int)((c + 255) / 256), 256, 0, stream>>>(pool_max, cnt_q, pool_pe, pool_bf);
  }
  {
    long c = (long)SK * D_DIM;
    finalize_emb<<<(int)((c + 255) / 256), 256, 0, stream>>>(emb_max, cnt_k, emb_c, emb_bf);
  }

  // projections
  int nblkQ = (SQ + 63) / 64;
  int nblkK = (SK + 63) / 64;
  gemm_bias<<<nblkQ, 256, 0, stream>>>(pool_bf, WTq, bq, nullptr, qbf, SQ);
  gemm_bias<<<nblkK, 256, 0, stream>>>(emb_bf, WTk, bk, nullptr, kbf, SK);
  gemm_bias<<<nblkK, 256, 0, stream>>>(emb_bf, WTv, bv, nullptr, vbf, SK);
  {
    long c = (long)SK * D_DIM;
    transpose_v<<<(int)((c + 255) / 256), 256, 0, stream>>>(vbf, vT);
  }

  // attention
  attn_kernel<<<BATCH * HEADS * QTILES, 32, 0, stream>>>(qbf, kbf, vT, cnt_k, attnbf);

  // output projection + residual + LN
  gemm_bias<<<nblkQ, 256, 0, stream>>>(attnbf, WTo, bo, o32, nullptr, SQ);
  residual_ln<<<SQ, 256, 0, stream>>>(pool_pe, o32, ln_g, ln_b, mhca);

  // gather back + seg head
  seg_head<<<N_PTS, 256, 0, stream>>>(features, mhca, seg_q, W_seg, b_seg, logits);
}
